// TransFuse_43499428773941
// MI455X (gfx1250) — matrix operand order repaired and checked by reading, not compiled, not run on hardware
//
#include <hip/hip_runtime.h>
#include <hip/hip_bf16.h>

typedef __attribute__((ext_vector_type(16))) _Float16 v16h;
typedef __attribute__((ext_vector_type(8)))  _Float16 v8h;
typedef __attribute__((ext_vector_type(8)))  float    v8f;

#define CDIM  128
#define NDIM  4096
#define BATCH 2
#define PADK  136   // 128 halves + 8 pad (272B row stride, 16B aligned)
#define NWAVE 4     // waves per block (split-K for attention)

// ---------------------------------------------------------------- WMMA core
__device__ __forceinline__ v8f wmma_f16(v16h a, v16h b, v8f c) {
  return __builtin_amdgcn_wmma_f32_16x16x32_f16(
      /*neg_a=*/false, a, /*neg_b=*/false, b,
      /*c_mod=*/(short)0, c, /*reuse_a=*/false, /*reuse_b=*/false);
}

// A fragment (16x32 f16). lane: row M = lane&15; elems 0..15 = K k0..k0+15,
// with k0 = kb*32 + 16*(lane>>4).
__device__ __forceinline__ v16h ldA_h(const _Float16* row, int k0) {
  v8h a = *(const v8h*)(row + k0);
  v8h b = *(const v8h*)(row + k0 + 8);
  v16h r;
#pragma unroll
  for (int i = 0; i < 8; ++i) { r[i] = a[i]; r[i + 8] = b[i]; }
  return r;
}

// B fragment (32x16 f16). lane: col N = lane&15; elems 0..15 = 16 contiguous
// K values starting at (kb*32 + 16*(lane>>4)). p pre-offset per lane.
__device__ __forceinline__ v16h ldB_h(const _Float16* p) {
  v8h a = *(const v8h*)p;
  v8h b = *(const v8h*)(p + 8);
  v16h r;
#pragma unroll
  for (int i = 0; i < 8; ++i) { r[i] = a[i]; r[i + 8] = b[i]; }
  return r;
}

// Row reductions across the 16 lanes sharing a C/D row (wave32: xor masks
// 1,2,4,8 keep the two half-wave groups separate).
__device__ __forceinline__ float redmax16(float v) {
#pragma unroll
  for (int m = 1; m <= 8; m <<= 1) v = fmaxf(v, __shfl_xor(v, m, 32));
  return v;
}
__device__ __forceinline__ float redsum16(float v) {
#pragma unroll
  for (int m = 1; m <= 8; m <<= 1) v += __shfl_xor(v, m, 32);
  return v;
}

// --------------------------------------------------- one-time prep kernels
// fp32 -> f16 bulk convert (weights). Each thread converts 8 elements.
__global__ __launch_bounds__(256)
void cvt_f16_kernel(const float* __restrict__ src, _Float16* __restrict__ dst) {
  const size_t idx = (size_t)blockIdx.x * blockDim.x + threadIdx.x;
  const float4* s = (const float4*)(src + idx * 8);
  float4 f0 = s[0], f1 = s[1];
  union { _Float16 h[8]; int4 u; } pk;
  pk.h[0] = (_Float16)f0.x; pk.h[1] = (_Float16)f0.y;
  pk.h[2] = (_Float16)f0.z; pk.h[3] = (_Float16)f0.w;
  pk.h[4] = (_Float16)f1.x; pk.h[5] = (_Float16)f1.y;
  pk.h[6] = (_Float16)f1.z; pk.h[7] = (_Float16)f1.w;
  ((int4*)dst)[idx] = pk.u;
}

// fp32 [B][C][N] -> f16 [B][N][C] transpose; 4 waves/block, one 16-wide
// n tile per wave, staged through per-wave LDS slices.
__global__ __launch_bounds__(32 * NWAVE)
void transpose_kernel(const float* __restrict__ src, _Float16* __restrict__ dst) {
  __shared__ _Float16 lds[NWAVE][16][PADK];
  const int lane  = threadIdx.x & 31;
  const int w     = threadIdx.x >> 5;
  const int tile  = blockIdx.x * NWAVE + w;
  const int batch = tile >> 8;
  const int n0    = (tile & 255) * 16;

#pragma unroll
  for (int rr = 0; rr < 4; ++rr) {
    const int k = lane + rr * 32;
    const float* p = src + ((size_t)batch * CDIM + k) * NDIM + n0;
#pragma unroll
    for (int jj = 0; jj < 4; ++jj) {
      float4 f = ((const float4*)p)[jj];
      lds[w][4*jj+0][k] = (_Float16)f.x; lds[w][4*jj+1][k] = (_Float16)f.y;
      lds[w][4*jj+2][k] = (_Float16)f.z; lds[w][4*jj+3][k] = (_Float16)f.w;
    }
  }
  // per-wave LDS slice; intra-wave DS ordering is sufficient (no barrier)
  const int n  = lane >> 1;          // 16 rows, 2 lanes each
  const int hh = (lane & 1) * 64;    // which 64-half half of the row
  _Float16* drow = dst + ((size_t)batch * NDIM + n0 + n) * CDIM + hh;
#pragma unroll
  for (int t = 0; t < 8; ++t)        // 8 x 16B
    *(int4*)(drow + t * 8) = *(const int4*)(&lds[w][n][hh + t * 8]);
}

// ------------------------------------------------------- QKV (1x1 conv) GEMM
// Pure f16 WMMA, no LDS. D[c, n] = relu(sum_k W[c,k]*src[k,n] + bias[c]).
// TRANSPOSED=true: store [n][c] f16 (Q, K).  false: store [c][n] f16 (V).
template <bool TRANSPOSED>
__device__ __forceinline__ void conv_gemm(const v16h bx[4],
                                          const _Float16* __restrict__ W,
                                          const float* __restrict__ bias,
                                          _Float16* __restrict__ dst,
                                          int batch, int n0, int lane) {
  const int lr = lane & 15, g = lane >> 4;
#pragma unroll
  for (int cc = 0; cc < 8; ++cc) {
    // Cluster the 8 b128 loads into distinct regs, then burst 4 WMMAs.
    v16h aw[4];
#pragma unroll
    for (int kb = 0; kb < 4; ++kb)
      aw[kb] = ldA_h(W + (size_t)(cc * 16 + lr) * CDIM, kb * 32 + 16 * g);
    v8f acc = {};
#pragma unroll
    for (int kb = 0; kb < 4; ++kb)
      acc = wmma_f16(aw[kb], bx[kb], acc);

    const int cbase = cc * 16 + 8 * g;
    if (TRANSPOSED) {  // compile-time: 8 c-contiguous halves -> one b128 store
      union { _Float16 h[8]; int4 u; } pk;
#pragma unroll
      for (int r = 0; r < 8; ++r)
        pk.h[r] = (_Float16)fmaxf(acc[r] + bias[cbase + r], 0.0f);
      *(int4*)(dst + ((size_t)batch * NDIM + n0 + lr) * CDIM + cbase) = pk.u;
    } else {
#pragma unroll
      for (int r = 0; r < 8; ++r)
        dst[((size_t)batch * CDIM + cbase + r) * NDIM + n0 + lr] =
            (_Float16)fmaxf(acc[r] + bias[cbase + r], 0.0f);
    }
  }
}

__global__ __launch_bounds__(32 * NWAVE)
void qkv_kernel(const _Float16* __restrict__ aT, const _Float16* __restrict__ bT,
                const _Float16* __restrict__ Wq, const float* __restrict__ bq,
                const _Float16* __restrict__ Wk, const float* __restrict__ bk,
                const _Float16* __restrict__ Wv, const float* __restrict__ bv,
                _Float16* __restrict__ Qh, _Float16* __restrict__ Kh,
                _Float16* __restrict__ Vh) {
  const int lane  = threadIdx.x & 31;
  const int w     = threadIdx.x >> 5;
  const int lr    = lane & 15, g = lane >> 4;
  const int tile  = blockIdx.x * NWAVE + w;  // one 16-col n tile per wave
  const int batch = tile >> 8;
  const int n0    = (tile & 255) * 16;

  // Hoist B fragments (one activation column tile feeds all 8 c-chunks).
  const _Float16* arow = aT + ((size_t)batch * NDIM + n0 + lr) * CDIM;
  const _Float16* brow = bT + ((size_t)batch * NDIM + n0 + lr) * CDIM;
  v16h ba[4], bb[4];
#pragma unroll
  for (int kb = 0; kb < 4; ++kb) {
    ba[kb] = ldB_h(arow + kb * 32 + 16 * g);
    bb[kb] = ldB_h(brow + kb * 32 + 16 * g);
  }

  conv_gemm<true >(ba, Wq, bq, Qh, batch, n0, lane);  // Q(a) -> [n][c]
  conv_gemm<true >(bb, Wk, bk, Kh, batch, n0, lane);  // K(b) -> [n][c]
  conv_gemm<false>(bb, Wv, bv, Vh, batch, n0, lane);  // V(b) -> [c][n]
}

// ---------------------------------------------- flash attention + out-proj
// 4 waves per block: each wave covers NDIM/4 keys with private online-softmax
// state; log-sum-exp combine in LDS; out-projection split across waves.
template <bool HAS_EXTRA, bool HAS_OUTT>
__global__ __launch_bounds__(32 * NWAVE)
void attn_kernel(const _Float16* __restrict__ Qh, const _Float16* __restrict__ Kh,
                 const _Float16* __restrict__ Vh,
                 const _Float16* __restrict__ Wo, const float* __restrict__ bo,
                 const float* __restrict__ resid, const float* __restrict__ extra,
                 float* __restrict__ out, _Float16* __restrict__ outT) {
  __shared__ _Float16 ldsP[NWAVE][16][32];     // per-wave P tile (D->A relayout)
  __shared__ float    ldsOp[NWAVE][16][CDIM];  // per-wave partial O (f32)
  __shared__ float    mstat[NWAVE][16], lstat[NWAVE][16];
  __shared__ _Float16 ldsO[16][PADK];          // combined O tile (f16)

  const int lane  = threadIdx.x & 31;
  const int w     = threadIdx.x >> 5;
  const int lr    = lane & 15, g = lane >> 4;
  const int bid   = blockIdx.x;
  const int batch = bid >> 8;
  const int n0    = (bid & 255) * 16;

  const _Float16* qrow = Qh + ((size_t)batch * NDIM + n0 + lr) * CDIM;
  v16h aq[4];
#pragma unroll
  for (int kb = 0; kb < 4; ++kb) aq[kb] = ldA_h(qrow, kb * 32 + 16 * g);

  const v8f vzero = {};
  v8f oacc[8];
#pragma unroll
  for (int i = 0; i < 8; ++i) oacc[i] = vzero;
  float mrow[8], lrow[8];
#pragma unroll
  for (int r = 0; r < 8; ++r) { mrow[r] = -3.0e38f; lrow[r] = 0.0f; }

  const _Float16* kmat = Kh + (size_t)batch * NDIM * CDIM;
  const _Float16* vmat = Vh + (size_t)batch * CDIM * NDIM;

  const int MT = NDIM / 32 / NWAVE;  // 32 key tiles per wave
  for (int i = 0; i < MT; ++i) {
    const int m0  = (w * MT + i) * 32;
    const int mpf = (w * MT + ((i + 1) & (MT - 1))) * 32;  // branch-free
    __builtin_prefetch(kmat + (size_t)(mpf + lr) * CDIM, 0, 0);

    // Cluster all 8 K-side B-frag loads, then burst 8 WMMAs.
    v16h bfr[8];
#pragma unroll
    for (int kb = 0; kb < 4; ++kb) {
      bfr[2*kb]   = ldB_h(kmat + (size_t)(m0 + lr)      * CDIM + kb * 32 + 16 * g);
      bfr[2*kb+1] = ldB_h(kmat + (size_t)(m0 + 16 + lr) * CDIM + kb * 32 + 16 * g);
    }
    v8f s0 = vzero, s1 = vzero;
#pragma unroll
    for (int kb = 0; kb < 4; ++kb) {
      s0 = wmma_f16(aq[kb], bfr[2*kb],   s0);
      s1 = wmma_f16(aq[kb], bfr[2*kb+1], s1);
    }

    // Online (unscaled) softmax; lane's slot r is row (8*g + r).
#pragma unroll
    for (int r = 0; r < 8; ++r) {
      float tmax = redmax16(fmaxf(s0[r], s1[r]));
      float nm   = fmaxf(mrow[r], tmax);
      float p0   = __expf(s0[r] - nm);
      float p1   = __expf(s1[r] - nm);
      float rs   = redsum16(p0 + p1);
      float corr = __expf(mrow[r] - nm);
      lrow[r] = lrow[r] * corr + rs;
      mrow[r] = nm;
#pragma unroll
      for (int cc = 0; cc < 8; ++cc) oacc[cc][r] *= corr;
      ldsP[w][8 * g + r][lr]      = (_Float16)p0;  // D -> A relayout via LDS
      ldsP[w][8 * g + r][16 + lr] = (_Float16)p1;
    }
    // per-wave LDS slice: intra-wave DS ordering suffices (no barrier)

    // Cluster all 8 V-side B-frag loads, then burst 8 WMMAs into O.
    v16h ap = ldA_h(&ldsP[w][lr][0], 16 * g);
#pragma unroll
    for (int cc = 0; cc < 8; ++cc)
      bfr[cc] = ldB_h(vmat + (size_t)(cc * 16 + lr) * NDIM + m0 + 16 * g);
#pragma unroll
    for (int cc = 0; cc < 8; ++cc)
      oacc[cc] = wmma_f16(ap, bfr[cc], oacc[cc]);
  }

  // Publish partial O + stats; combine across the 4 waves (log-sum-exp).
#pragma unroll
  for (int r = 0; r < 8; ++r) {
#pragma unroll
    for (int cc = 0; cc < 8; ++cc)
      ldsOp[w][8 * g + r][cc * 16 + lr] = oacc[cc][r];
  }
  if (lr == 0) {
#pragma unroll
    for (int r = 0; r < 8; ++r) {
      mstat[w][8 * g + r] = mrow[r];
      lstat[w][8 * g + r] = lrow[r];
    }
  }
  __syncthreads();

  {
    const int t   = threadIdx.x;       // 128 threads, 16 elems each
    const int row = t >> 3;
    const int cb  = (t & 7) * 16;
    float ms = mstat[0][row];
#pragma unroll
    for (int u = 1; u < NWAVE; ++u) ms = fmaxf(ms, mstat[u][row]);
    float scale[NWAVE], lsum = 0.0f;
#pragma unroll
    for (int u = 0; u < NWAVE; ++u) {
      scale[u] = __expf(mstat[u][row] - ms);
      lsum += lstat[u][row] * scale[u];
    }
    const float inv = 1.0f / lsum;
#pragma unroll
    for (int j = 0; j < 16; ++j) {
      float v = 0.0f;
#pragma unroll
      for (int u = 0; u < NWAVE; ++u) v += ldsOp[u][row][cb + j] * scale[u];
      ldsO[row][cb + j] = (_Float16)(v * inv);
    }
  }
  __syncthreads();

  // Output projection: D[n, o] = relu(O x Wo^T + bo) + resid (+ extra).
  // Split the 8 output-channel chunks across the 4 waves (2 each).
  v16h ao[4];
#pragma unroll
  for (int kb = 0; kb < 4; ++kb) ao[kb] = ldA_h(&ldsO[lr][0], kb * 32 + 16 * g);

#pragma unroll
  for (int oi = 0; oi < 2; ++oi) {
    const int oc = w * 2 + oi;
    v16h bw[4];
#pragma unroll
    for (int kb = 0; kb < 4; ++kb)
      bw[kb] = ldB_h(Wo + (size_t)(oc * 16 + lr) * CDIM + kb * 32 + 16 * g);
    v8f acc = vzero;
#pragma unroll
    for (int kb = 0; kb < 4; ++kb)
      acc = wmma_f16(ao[kb], bw[kb], acc);

    const int o = oc * 16 + lr;
    const float bb = bo[o];
    const size_t base = ((size_t)batch * CDIM + o) * NDIM + n0 + 8 * g;
    float vals[8];
#pragma unroll
    for (int r = 0; r < 8; ++r) {
      float v = fmaxf(acc[r] + bb, 0.0f) + resid[base + r];
      if (HAS_EXTRA) v += extra[base + r];   // compile-time
      vals[r] = v;
    }
    *(float4*)(out + base)     = make_float4(vals[0], vals[1], vals[2], vals[3]);
    *(float4*)(out + base + 4) = make_float4(vals[4], vals[5], vals[6], vals[7]);
    if (HAS_OUTT) {                          // compile-time
#pragma unroll
      for (int r = 0; r < 8; ++r)
        outT[((size_t)batch * NDIM + n0 + 8 * g + r) * CDIM + o] =
            (_Float16)vals[r];
    }
  }
}

// ------------------------------------------------------------------ launcher
extern "C" void kernel_launch(void* const* d_in, const int* in_sizes, int n_in,
                              void* d_out, int out_size, void* d_ws, size_t ws_size,
                              hipStream_t stream) {
  (void)in_sizes; (void)n_in; (void)out_size; (void)ws_size;
  const float* x  = (const float*)d_in[0];
  const float* y  = (const float*)d_in[1];
  const float* Wq = (const float*)d_in[2];
  const float* bq = (const float*)d_in[3];
  const float* Wk = (const float*)d_in[4];
  const float* bk = (const float*)d_in[5];
  const float* Wv = (const float*)d_in[6];
  const float* bv = (const float*)d_in[7];
  const float* Wo = (const float*)d_in[8];
  const float* bo = (const float*)d_in[9];
  float* out = (float*)d_out;

  const size_t BNC = (size_t)BATCH * NDIM * CDIM;  // 1,048,576 elements
  const size_t WSZ = (size_t)4 * CDIM * CDIM;      // 65,536 per weight stack
  _Float16* Wqh = (_Float16*)d_ws;
  _Float16* Wkh = Wqh + WSZ;
  _Float16* Wvh = Wkh + WSZ;
  _Float16* Woh = Wvh + WSZ;
  _Float16* Qh  = Woh + WSZ;
  _Float16* Kh  = Qh + BNC;
  _Float16* Vh  = Kh + BNC;
  _Float16* xT  = Vh + BNC;
  _Float16* yT  = xT + BNC;
  _Float16* t0T = yT + BNC;
  _Float16* t1T = t0T + BNC;
  float* t0 = (float*)(t1T + BNC);
  float* t1 = t0 + BNC;
  float* t2 = t1 + BNC;

  const int NTILE = BATCH * (NDIM / 16);      // 512 16-row tiles
  dim3 tgrid(NTILE / NWAVE), tblock(32 * NWAVE);  // 128 blocks x 4 waves
  dim3 agrid(NTILE), ablock(32 * NWAVE);          // 512 blocks x 4 waves

  // One-time prep: f16 weights + transposed f16 activations.
  cvt_f16_kernel<<<dim3(WSZ / (256 * 8)), dim3(256), 0, stream>>>(Wq, Wqh);
  cvt_f16_kernel<<<dim3(WSZ / (256 * 8)), dim3(256), 0, stream>>>(Wk, Wkh);
  cvt_f16_kernel<<<dim3(WSZ / (256 * 8)), dim3(256), 0, stream>>>(Wv, Wvh);
  cvt_f16_kernel<<<dim3(WSZ / (256 * 8)), dim3(256), 0, stream>>>(Wo, Woh);
  transpose_kernel<<<tgrid, tblock, 0, stream>>>(x, xT);
  transpose_kernel<<<tgrid, tblock, 0, stream>>>(y, yT);

  auto W = [&](_Float16* w, int i) { return w + (size_t)i * CDIM * CDIM; };
  auto Bx = [&](const float* w, int i) { return w + (size_t)i * CDIM; };

  // layer 0: fuse_xy = trans(x, y)
  qkv_kernel<<<tgrid, tblock, 0, stream>>>(xT, yT, W(Wqh,0), Bx(bq,0), W(Wkh,0), Bx(bk,0),
                                           W(Wvh,0), Bx(bv,0), Qh, Kh, Vh);
  attn_kernel<false, true><<<agrid, ablock, 0, stream>>>(
      Qh, Kh, Vh, W(Woh,0), Bx(bo,0), x, nullptr, t0, t0T);
  // layer 1: fuse_yz = trans(y, x)
  qkv_kernel<<<tgrid, tblock, 0, stream>>>(yT, xT, W(Wqh,1), Bx(bq,1), W(Wkh,1), Bx(bk,1),
                                           W(Wvh,1), Bx(bv,1), Qh, Kh, Vh);
  attn_kernel<false, true><<<agrid, ablock, 0, stream>>>(
      Qh, Kh, Vh, W(Woh,1), Bx(bo,1), y, nullptr, t1, t1T);
  // layer 2: fuse_xx = trans(t0, t0)
  qkv_kernel<<<tgrid, tblock, 0, stream>>>(t0T, t0T, W(Wqh,2), Bx(bq,2), W(Wkh,2), Bx(bk,2),
                                           W(Wvh,2), Bx(bv,2), Qh, Kh, Vh);
  attn_kernel<false, false><<<agrid, ablock, 0, stream>>>(
      Qh, Kh, Vh, W(Woh,2), Bx(bo,2), t0, nullptr, t2, nullptr);
  // layer 3: fuse_yy = trans(t1, t1); out = fuse_yy + fuse_xx
  qkv_kernel<<<tgrid, tblock, 0, stream>>>(t1T, t1T, W(Wqh,3), Bx(bq,3), W(Wkh,3), Bx(bk,3),
                                           W(Wvh,3), Bx(bv,3), Qh, Kh, Vh);
  attn_kernel<true, false><<<agrid, ablock, 0, stream>>>(
      Qh, Kh, Vh, W(Woh,3), Bx(bo,3), t1, t2, out, nullptr);
}
